// BottomUpSummarizer_88450556494032
// MI455X (gfx1250) — matrix operand order repaired and checked by reading, not compiled, not run on hardware
//
#include <hip/hip_runtime.h>
#include <hip/hip_bf16.h>

// ---------------------------------------------------------------------------
// BottomUpSummarizer on gfx1250 (MI455X): WMMA f16->f32 for every GEMM,
// persistent-LDS recurrent LSTM kernels, step-invariant attention hoisting.
// Round 2: EXEC-divergence-free A-fragment path for full M tiles.
// ---------------------------------------------------------------------------

typedef __attribute__((ext_vector_type(16))) _Float16 v16h;
typedef __attribute__((ext_vector_type(8)))  float    v8f;

#define Bc 8
#define Sc 200
#define Tc 51
#define Hc 256
#define Vc 32000
#define EVc 64000
#define Tm1 50

static __device__ __forceinline__ float sigmoidf_(float x) {
    return 1.0f / (1.0f + __expf(-x));
}

// ---- WMMA fragment loaders (CDNA5 16x16x32 f16 layouts, wave32) -----------
// A (16x32, MxK): lanes 0-15 hold M=lane, K in {0..7,16..23}; lanes 16-31 hold
// M=lane-16, K in {8..15,24..31}.
static __device__ __forceinline__ v16h load_a_frag_full(const _Float16* A, int lda,
                                                        int lane, int k0) {
    int m  = lane & 15;
    int kg = (lane >> 4) << 3;   // 0 or 8
    const _Float16* p = A + (size_t)m * lda + k0 + kg;
    v16h a;
#pragma unroll
    for (int i = 0; i < 8; ++i) a[i] = p[i];
#pragma unroll
    for (int i = 0; i < 8; ++i) a[8 + i] = p[16 + i];
    return a;
}

static __device__ __forceinline__ v16h load_a_frag_pred(const _Float16* A, int lda,
                                                        int mRows, int lane, int k0) {
    int m  = lane & 15;
    int kg = (lane >> 4) << 3;   // 0 or 8
    v16h a = {};
    if (m < mRows) {
        const _Float16* p = A + (size_t)m * lda + k0 + kg;
#pragma unroll
        for (int i = 0; i < 8; ++i) a[i] = p[i];
#pragma unroll
        for (int i = 0; i < 8; ++i) a[8 + i] = p[16 + i];
    }
    return a;
}

// B (32x16, KxN) taken from row-major W (N x K) so the WMMA computes A @ W^T.
// lanes 0-15: N=lane, K in {0..7,16..23}; lanes 16-31: N=lane-16, K in {8..15,24..31}.
static __device__ __forceinline__ v16h load_b_frag(const _Float16* W, int ldw,
                                                   int lane, int n0, int k0) {
    int n  = lane & 15;
    int kg = (lane >> 4) << 3;   // 0 or 8
    const _Float16* p = W + (size_t)(n0 + n) * ldw + k0 + kg;
    v16h b;
#pragma unroll
    for (int j = 0; j < 8; ++j) { b[j] = p[j]; b[8 + j] = p[16 + j]; }
    return b;
}

// ---- Generic C = A(MxK) @ W(NxK)^T  (+bias, optional tanh, optional +=) ----
// FULLM: every M tile has 16 valid rows (M % 16 == 0) -> no EXEC juggling in
// the WMMA K-loop.
template <bool FULLM>
__global__ void gemm_wmma_kernel(const _Float16* __restrict__ A, int lda,
                                 const _Float16* __restrict__ W, int ldw,
                                 const float* __restrict__ bias,
                                 float* __restrict__ C, int ldc,
                                 int M, int N, int K, int act, int accum) {
    const int lane = threadIdx.x & 31;
    const int wavesPerBlock = blockDim.x >> 5;
    const int gwave = blockIdx.x * wavesPerBlock + (threadIdx.x >> 5);
    const int tilesN = N >> 4;
    const int tilesM = (M + 15) >> 4;
    if (gwave >= tilesM * tilesN) return;
    const int tm = gwave / tilesN;
    const int tn = gwave - tm * tilesN;
    const int m0 = tm << 4, n0 = tn << 4;
    const int mRows = FULLM ? 16 : ((M - m0 < 16) ? (M - m0) : 16);
    v8f acc = {};
    for (int k0 = 0; k0 < K; k0 += 32) {
        v16h a = FULLM ? load_a_frag_full(A + (size_t)m0 * lda, lda, lane, k0)
                       : load_a_frag_pred(A + (size_t)m0 * lda, lda, mRows, lane, k0);
        v16h b = load_b_frag(W, ldw, lane, n0, k0);
        acc = __builtin_amdgcn_wmma_f32_16x16x32_f16(false, a, false, b,
                                                     (short)0, acc, false, false);
    }
    const int n = lane & 15;
    const int mBase = (lane >> 4) << 3;
    const float bv = bias ? bias[n0 + n] : 0.0f;
#pragma unroll
    for (int r = 0; r < 8; ++r) {
        int m = mBase + r;
        if (FULLM || m < mRows) {
            float val = acc[r] + bv;
            if (act == 1) val = tanhf(val);
            float* p = C + (size_t)(m0 + m) * ldc + (n0 + n);
            if (accum) *p += val; else *p = val;
        }
    }
}

// ---- helpers --------------------------------------------------------------
__global__ void f2h_kernel(const float* __restrict__ src,
                           _Float16* __restrict__ dst, int n) {
    int i = blockIdx.x * blockDim.x + threadIdx.x;
    if (i < n) dst[i] = (_Float16)src[i];
}

// emb_h layout: [s][b][h]  (row = s*B + b), f16 — A-matrix for xproj GEMMs.
__global__ void embed_kernel(const int* __restrict__ x,
                             const float* __restrict__ emb_enc,
                             _Float16* __restrict__ emb_h) {
    int i = blockIdx.x * blockDim.x + threadIdx.x;   // S*B*H
    if (i >= Sc * Bc * Hc) return;
    int h  = i & (Hc - 1);
    int sb = i >> 8;
    int b  = sb & 7;
    int s  = sb >> 3;
    int tok = x[b * Sc + s];
    emb_h[i] = (_Float16)emb_enc[(size_t)tok * Hc + h];
}

__global__ void init_kernel(float* cov, float* lossacc, _Float16* hdec_h) {
    int i = blockIdx.x * blockDim.x + threadIdx.x;
    if (i < Bc * Sc) cov[i] = 0.0f;
    if (i < 8)       lossacc[i] = 0.0f;
    if (i < 16 * Hc) hdec_h[i] = (_Float16)0.0f;
}

// ---- Persistent bidirectional LSTM: one block per direction ---------------
// z_t = xproj[t] + h_{t-1} @ whh^T + bih + bhh  (h in LDS as f16, WMMA).
__global__ void bilstm_kernel(const float* __restrict__ xproj_f,
                              const float* __restrict__ xproj_b,
                              const _Float16* __restrict__ whhf_h,
                              const _Float16* __restrict__ whhb_h,
                              const float* __restrict__ bih_f, const float* __restrict__ bhh_f,
                              const float* __restrict__ bih_b, const float* __restrict__ bhh_b,
                              float* __restrict__ enc_out,
                              _Float16* __restrict__ enc_out_h,
                              float* __restrict__ hT, float* __restrict__ cT) {
    const int dir = blockIdx.x;
    const float*    xproj = dir ? xproj_b : xproj_f;
    const _Float16* whh   = dir ? whhb_h  : whhf_h;
    const float*    bi    = dir ? bih_b   : bih_f;
    const float*    bh    = dir ? bhh_b   : bhh_f;

    __shared__ _Float16 hS[16 * Hc];        // rows 8..15 stay zero (M pad)
    __shared__ float    cS[Bc * Hc];
    __shared__ float    zS[Bc * 4 * Hc];

    const int tid = threadIdx.x;            // 256 threads = 8 waves
    for (int i = tid; i < 16 * Hc; i += 256) hS[i] = (_Float16)0.0f;
    for (int i = tid; i < Bc * Hc; i += 256) cS[i] = 0.0f;
    __syncthreads();

    const int lane = tid & 31;
    const int wave = tid >> 5;              // 0..7, each owns 8 N-tiles

    for (int step = 0; step < Sc; ++step) {
        const int s_src = dir ? (Sc - 1 - step) : step;
        for (int j = 0; j < 8; ++j) {
            const int n0 = (wave * 8 + j) << 4;
            v8f acc = {};
            for (int k0 = 0; k0 < Hc; k0 += 32) {
                v16h a = load_a_frag_full(hS, Hc, lane, k0);
                v16h b = load_b_frag(whh, Hc, lane, n0, k0);
                acc = __builtin_amdgcn_wmma_f32_16x16x32_f16(false, a, false, b,
                                                             (short)0, acc, false, false);
            }
            const int n = lane & 15;
            const int mBase = (lane >> 4) << 3;
#pragma unroll
            for (int r = 0; r < 8; ++r) {
                int m = mBase + r;
                if (m < Bc) {
                    int col = n0 + n;
                    zS[m * 1024 + col] = acc[r]
                        + xproj[((size_t)s_src * Bc + m) * 1024 + col]
                        + bi[col] + bh[col];
                }
            }
        }
        __syncthreads();
        for (int e = tid; e < Bc * Hc; e += 256) {
            int b = e >> 8, jj = e & (Hc - 1);
            float zi = zS[b * 1024 + jj];
            float zf = zS[b * 1024 + 256 + jj];
            float zg = zS[b * 1024 + 512 + jj];
            float zo = zS[b * 1024 + 768 + jj];
            float c = sigmoidf_(zf) * cS[e] + sigmoidf_(zi) * tanhf(zg);
            float h = sigmoidf_(zo) * tanhf(c);
            cS[e] = c;
            hS[b * Hc + jj] = (_Float16)h;
            size_t o = ((size_t)b * Sc + s_src) * (2 * Hc) + dir * Hc + jj;
            enc_out[o]   = h;
            enc_out_h[o] = (_Float16)h;
        }
        __syncthreads();
    }
    for (int e = tid; e < Bc * Hc; e += 256) {
        int b = e >> 8, jj = e & (Hc - 1);
        hT[dir * (Bc * Hc) + e] = (float)hS[b * Hc + jj];
        cT[dir * (Bc * Hc) + e] = cS[e];
    }
}

// ---- h0 / c0 (relu of concat @ W^T), tiny ---------------------------------
__global__ void h0c0_kernel(const float* __restrict__ hT, const float* __restrict__ cT,
                            const float* __restrict__ w_rh, const float* __restrict__ b_rh,
                            const float* __restrict__ w_rc, const float* __restrict__ b_rc,
                            float* __restrict__ hdec_f, _Float16* __restrict__ hdec_h,
                            float* __restrict__ cdec) {
    int idx = blockIdx.x * blockDim.x + threadIdx.x;
    if (idx >= Bc * Hc) return;
    int b = idx >> 8, j = idx & (Hc - 1);
    float ah = b_rh[j], ac = b_rc[j];
    for (int k = 0; k < Hc; ++k) {
        float hf = hT[b * Hc + k], hb = hT[Bc * Hc + b * Hc + k];
        ah += hf * w_rh[j * 512 + k] + hb * w_rh[j * 512 + 256 + k];
        float cf = cT[b * Hc + k], cb = cT[Bc * Hc + b * Hc + k];
        ac += cf * w_rc[j * 512 + k] + cb * w_rc[j * 512 + 256 + k];
    }
    ah = ah > 0.0f ? ah : 0.0f;
    ac = ac > 0.0f ? ac : 0.0f;
    hdec_f[idx] = ah;
    hdec_h[b * Hc + j] = (_Float16)ah;
    cdec[idx] = ac;
}

// ---- copy_probs + BCE loss accumulation -----------------------------------
__global__ void copyprobs_kernel(const float* __restrict__ sel1,
                                 const float* __restrict__ w_sel2,
                                 const float* __restrict__ b_sel2,
                                 const int* __restrict__ x,
                                 float* __restrict__ cp, float* __restrict__ maskcp,
                                 float* __restrict__ lossacc) {
    int i = blockIdx.x * blockDim.x + threadIdx.x;   // B*S
    if (i >= Bc * Sc) return;
    float a = b_sel2[0];
    const float* r = sel1 + (size_t)i * Hc;
    for (int k = 0; k < Hc; ++k) a += r[k] * w_sel2[k];
    float p = sigmoidf_(a);
    cp[i] = p;
    maskcp[i] = (p > 1e-8f) ? 1.0f : 0.0f;
    float m = (x[i] != 0) ? 1.0f : 0.0f;
    float lp  = logf(p);       if (lp  < -100.0f) lp  = -100.0f;
    float l1p = log1pf(-p);    if (l1p < -100.0f) l1p = -100.0f;
    float bce = -(m * lp + (1.0f - m) * l1p);
    atomicAdd(&lossacc[2], bce * m);
    atomicAdd(&lossacc[3], m);
}

// ---- rank-1 coverage factorization: u = Wa3 @ w_cov, v = Wa3 @ b_cov ------
__global__ void uv_kernel(const float* __restrict__ w_attn,
                          const float* __restrict__ w_cov,
                          const float* __restrict__ b_cov,
                          float* __restrict__ u, float* __restrict__ v) {
    int j = blockIdx.x * blockDim.x + threadIdx.x;
    if (j >= Hc) return;
    float su = 0.0f, sv = 0.0f;
    for (int h = 0; h < Hc; ++h) {
        float wa = w_attn[j * 1024 + 768 + h];
        su += wa * w_cov[h];
        sv += wa * b_cov[h];
    }
    u[j] = su; v[j] = sv;
}

// ---- per-step attention: logits, masked softmax, coverage, context --------
__global__ void attn_kernel(const float* __restrict__ ebase,
                            const float* __restrict__ hWa2,
                            const float* __restrict__ u, const float* __restrict__ v,
                            const float* __restrict__ b_attn,
                            const float* __restrict__ w_attn_v,
                            const float* __restrict__ maskcp,
                            const float* __restrict__ enc_out,
                            float* __restrict__ cov, float* __restrict__ attnw,
                            float* __restrict__ ctx) {
    const int b = blockIdx.x;
    const int tid = threadIdx.x;
    __shared__ float logitS[Sc];
    __shared__ float red[256];
    if (tid < Sc) {
        float cv = cov[b * Sc + tid];
        const float* eb = ebase + ((size_t)b * Sc + tid) * Hc;
        const float* hw = hWa2 + b * Hc;
        float acc = 0.0f;
        for (int j = 0; j < Hc; ++j) {
            float e = tanhf(eb[j] + hw[j] + cv * u[j] + v[j] + b_attn[j]);
            acc += e * w_attn_v[j];
        }
        logitS[tid] = acc;
    }
    __syncthreads();
    float mval = (tid < Sc) ? logitS[tid] : -1e30f;
    red[tid] = mval; __syncthreads();
    for (int off = 128; off > 0; off >>= 1) { if (tid < off) red[tid] = fmaxf(red[tid], red[tid + off]); __syncthreads(); }
    float gmax = red[0]; __syncthreads();
    float s = 0.0f;
    if (tid < Sc) { float e = __expf(logitS[tid] - gmax); logitS[tid] = e; s = e; }
    red[tid] = s; __syncthreads();
    for (int off = 128; off > 0; off >>= 1) { if (tid < off) red[tid] += red[tid + off]; __syncthreads(); }
    float gsum = red[0]; __syncthreads();
    float a2 = 0.0f;
    if (tid < Sc) { float a = (logitS[tid] / gsum) * maskcp[b * Sc + tid]; logitS[tid] = a; a2 = a; }
    red[tid] = a2; __syncthreads();
    for (int off = 128; off > 0; off >>= 1) { if (tid < off) red[tid] += red[tid + off]; __syncthreads(); }
    float asum = red[0] + 1e-10f; __syncthreads();
    if (tid < Sc) {
        float a = logitS[tid] / asum;
        logitS[tid] = a;
        attnw[b * Sc + tid] = a;
        cov[b * Sc + tid] += a;
    }
    __syncthreads();
    for (int j = tid; j < 2 * Hc; j += 256) {
        float s2 = 0.0f;
        for (int s0 = 0; s0 < Sc; ++s0)
            s2 += logitS[s0] * enc_out[((size_t)b * Sc + s0) * (2 * Hc) + j];
        ctx[b * (2 * Hc) + j] = s2;
    }
}

// ---- build f16 decoder input [emb_dec[tok] | ctx], padded to 16 rows ------
__global__ void decin_kernel(const int* __restrict__ labels,
                             const float* __restrict__ emb_dec,
                             const float* __restrict__ ctx,
                             _Float16* __restrict__ decin, int t) {
    int i = blockIdx.x * blockDim.x + threadIdx.x;   // 16*768
    if (i >= 16 * 768) return;
    int m = i / 768, col = i % 768;
    float val = 0.0f;
    if (m < Bc) {
        if (col < Hc) {
            int tok = (t == 0) ? 1 : labels[m * Tc + t];
            val = emb_dec[(size_t)tok * Hc + col];
        } else {
            val = ctx[m * (2 * Hc) + (col - Hc)];
        }
    }
    decin[i] = (_Float16)val;
}

__global__ void deccell_kernel(const float* __restrict__ zdec,
                               const float* __restrict__ bih_d,
                               const float* __restrict__ bhh_d,
                               float* __restrict__ hdec_f,
                               _Float16* __restrict__ hdec_h,
                               float* __restrict__ cdec) {
    int idx = blockIdx.x * blockDim.x + threadIdx.x;
    if (idx >= Bc * Hc) return;
    int b = idx >> 8, j = idx & (Hc - 1);
    float zi = zdec[b * 1024 + j]        + bih_d[j]        + bhh_d[j];
    float zf = zdec[b * 1024 + 256 + j]  + bih_d[256 + j]  + bhh_d[256 + j];
    float zg = zdec[b * 1024 + 512 + j]  + bih_d[512 + j]  + bhh_d[512 + j];
    float zo = zdec[b * 1024 + 768 + j]  + bih_d[768 + j]  + bhh_d[768 + j];
    float c = sigmoidf_(zf) * cdec[idx] + sigmoidf_(zi) * tanhf(zg);
    float h = sigmoidf_(zo) * tanhf(c);
    cdec[idx] = c;
    hdec_f[idx] = h;
    hdec_h[b * Hc + j] = (_Float16)h;
}

__global__ void pgen_kernel(const float* __restrict__ ctx,
                            const float* __restrict__ hdec_f,
                            const float* __restrict__ w_ptr,
                            const float* __restrict__ b_ptr,
                            float* __restrict__ pgen) {
    int b = threadIdx.x;
    if (b >= Bc) return;
    float a = b_ptr[0];
    for (int k = 0; k < 2 * Hc; ++k) a += ctx[b * (2 * Hc) + k] * w_ptr[k];
    for (int k = 0; k < Hc; ++k)     a += hdec_f[b * Hc + k] * w_ptr[2 * Hc + k];
    pgen[b] = sigmoidf_(a);
}

// ---- vocab softmax, write p_gen * vocab_dist, zero extended half ----------
__global__ void vocab_write_kernel(const float* __restrict__ vlogits,
                                   const float* __restrict__ pgen,
                                   float* __restrict__ out, int t) {
    const int b = blockIdx.x;
    const int tid = threadIdx.x;
    __shared__ float red[256];
    const float* row = vlogits + (size_t)b * Vc;
    float mval = -1e30f;
    for (int n = tid; n < Vc; n += 256) mval = fmaxf(mval, row[n]);
    red[tid] = mval; __syncthreads();
    for (int off = 128; off > 0; off >>= 1) { if (tid < off) red[tid] = fmaxf(red[tid], red[tid + off]); __syncthreads(); }
    float gmax = red[0]; __syncthreads();
    float s = 0.0f;
    for (int n = tid; n < Vc; n += 256) s += __expf(row[n] - gmax);
    red[tid] = s; __syncthreads();
    for (int off = 128; off > 0; off >>= 1) { if (tid < off) red[tid] += red[tid + off]; __syncthreads(); }
    float inv = pgen[b] / red[0];
    float* orow = out + ((size_t)b * Tm1 + t) * EVc;
    for (int n = tid; n < Vc; n += 256)       orow[n] = __expf(row[n] - gmax) * inv;
    for (int n = Vc + tid; n < EVc; n += 256) orow[n] = 0.0f;
}

__global__ void scatter_kernel(const int* __restrict__ x,
                               const float* __restrict__ attnw,
                               const float* __restrict__ pgen,
                               float* __restrict__ out, int t) {
    int i = blockIdx.x * blockDim.x + threadIdx.x;   // B*S
    if (i >= Bc * Sc) return;
    int b = i / Sc;
    float add = (1.0f - pgen[b]) * attnw[i];
    atomicAdd(&out[((size_t)b * Tm1 + t) * EVc + x[i]], add);
}

// ---- CE loss over 64000-wide rows -----------------------------------------
__global__ void celoss_kernel(const float* __restrict__ out,
                              const int* __restrict__ labels,
                              float* __restrict__ lossacc) {
    const int row = blockIdx.x;               // b*Tm1 + t
    const int b = row / Tm1, t = row % Tm1;
    const int tid = threadIdx.x;
    __shared__ float red[256];
    const float* r = out + (size_t)row * EVc;
    float mval = -1e30f;
    for (int n = tid; n < EVc; n += 256) mval = fmaxf(mval, r[n]);
    red[tid] = mval; __syncthreads();
    for (int off = 128; off > 0; off >>= 1) { if (tid < off) red[tid] = fmaxf(red[tid], red[tid + off]); __syncthreads(); }
    float gmax = red[0]; __syncthreads();
    float s = 0.0f;
    for (int n = tid; n < EVc; n += 256) s += __expf(r[n] - gmax);
    red[tid] = s; __syncthreads();
    for (int off = 128; off > 0; off >>= 1) { if (tid < off) red[tid] += red[tid + off]; __syncthreads(); }
    if (tid == 0) {
        int tgt = labels[b * Tc + t + 1];
        float valid = (tgt != 0) ? 1.0f : 0.0f;
        float nll = -(r[tgt] - gmax - logf(red[0]));
        atomicAdd(&lossacc[0], nll * valid);
        atomicAdd(&lossacc[1], valid);
    }
}

__global__ void finalloss_kernel(const float* __restrict__ lossacc,
                                 float* __restrict__ out, int out_size) {
    if (threadIdx.x == 0 && blockIdx.x == 0)
        out[out_size - 1] = lossacc[0] / lossacc[1] + lossacc[2] / lossacc[3];
}

// ===========================================================================
extern "C" void kernel_launch(void* const* d_in, const int* in_sizes, int n_in,
                              void* d_out, int out_size, void* d_ws, size_t ws_size,
                              hipStream_t stream) {
    (void)in_sizes; (void)n_in; (void)ws_size;
    const int* x        = (const int*)d_in[0];
    const int* labels   = (const int*)d_in[1];
    const float* emb_enc = (const float*)d_in[3];
    const float* wih_f = (const float*)d_in[4],  *whh_f = (const float*)d_in[5];
    const float* bih_f = (const float*)d_in[6],  *bhh_f = (const float*)d_in[7];
    const float* wih_b = (const float*)d_in[8],  *whh_b = (const float*)d_in[9];
    const float* bih_b = (const float*)d_in[10], *bhh_b = (const float*)d_in[11];
    const float* w_sel1 = (const float*)d_in[12], *b_sel1 = (const float*)d_in[13];
    const float* w_sel2 = (const float*)d_in[14], *b_sel2 = (const float*)d_in[15];
    const float* w_rh = (const float*)d_in[16], *b_rh = (const float*)d_in[17];
    const float* w_rc = (const float*)d_in[18], *b_rc = (const float*)d_in[19];
    const float* emb_dec = (const float*)d_in[20];
    const float* wih_d = (const float*)d_in[21], *whh_d = (const float*)d_in[22];
    const float* bih_d = (const float*)d_in[23], *bhh_d = (const float*)d_in[24];
    const float* w_attn = (const float*)d_in[25], *b_attn = (const float*)d_in[26];
    const float* w_attn_v = (const float*)d_in[27];
    const float* w_ptr = (const float*)d_in[28], *b_ptr = (const float*)d_in[29];
    const float* w_gen = (const float*)d_in[30], *b_gen = (const float*)d_in[31];
    const float* w_cov = (const float*)d_in[32], *b_cov = (const float*)d_in[33];
    float* out = (float*)d_out;

    // ---- workspace bump allocator ----
    size_t off = 0;
    auto alloc = [&](size_t bytes) -> void* {
        off = (off + 255) & ~((size_t)255);
        void* p = (char*)d_ws + off;
        off += bytes;
        return p;
    };
    _Float16* emb_h    = (_Float16*)alloc((size_t)Sc * Bc * Hc * 2);
    _Float16* encout_h = (_Float16*)alloc((size_t)Bc * Sc * 2 * Hc * 2);
    _Float16* wihf_h   = (_Float16*)alloc((size_t)1024 * 256 * 2);
    _Float16* whhf_h   = (_Float16*)alloc((size_t)1024 * 256 * 2);
    _Float16* wihb_h   = (_Float16*)alloc((size_t)1024 * 256 * 2);
    _Float16* whhb_h   = (_Float16*)alloc((size_t)1024 * 256 * 2);
    _Float16* wsel1_h  = (_Float16*)alloc((size_t)256 * 512 * 2);
    _Float16* wattn_h  = (_Float16*)alloc((size_t)256 * 1024 * 2);
    _Float16* wihd_h   = (_Float16*)alloc((size_t)1024 * 768 * 2);
    _Float16* whhd_h   = (_Float16*)alloc((size_t)1024 * 256 * 2);
    _Float16* wgen_h   = (_Float16*)alloc((size_t)Vc * Hc * 2);
    _Float16* hdec_h   = (_Float16*)alloc((size_t)16 * Hc * 2);
    _Float16* decin_h  = (_Float16*)alloc((size_t)16 * 768 * 2);
    float* xprojf  = (float*)alloc((size_t)Sc * Bc * 1024 * 4);
    float* xprojb  = (float*)alloc((size_t)Sc * Bc * 1024 * 4);
    float* encout  = (float*)alloc((size_t)Bc * Sc * 2 * Hc * 4);
    float* hTbuf   = (float*)alloc((size_t)2 * Bc * Hc * 4);
    float* cTbuf   = (float*)alloc((size_t)2 * Bc * Hc * 4);
    float* sel1    = (float*)alloc((size_t)Bc * Sc * Hc * 4);
    float* cp      = (float*)alloc((size_t)Bc * Sc * 4);
    float* maskcp  = (float*)alloc((size_t)Bc * Sc * 4);
    float* ebase   = (float*)alloc((size_t)Bc * Sc * Hc * 4);
    float* uvec    = (float*)alloc((size_t)Hc * 4);
    float* vvec    = (float*)alloc((size_t)Hc * 4);
    float* hWa2    = (float*)alloc((size_t)Bc * Hc * 4);
    float* cov     = (float*)alloc((size_t)Bc * Sc * 4);
    float* attnw   = (float*)alloc((size_t)Bc * Sc * 4);
    float* ctx     = (float*)alloc((size_t)Bc * 2 * Hc * 4);
    float* hdec_f  = (float*)alloc((size_t)Bc * Hc * 4);
    float* cdec    = (float*)alloc((size_t)Bc * Hc * 4);
    float* zdec    = (float*)alloc((size_t)16 * 1024 * 4);
    float* pgen    = (float*)alloc((size_t)Bc * 4);
    float* vlogits = (float*)alloc((size_t)Bc * Vc * 4);
    float* lossacc = (float*)alloc((size_t)8 * 4);

    auto gemm = [&](const _Float16* A, int lda, const _Float16* W, int ldw,
                    const float* bias, float* C, int ldc,
                    int M, int N, int K, int act, int accum) {
        int tiles = ((M + 15) / 16) * (N / 16);
        int blocks = (tiles + 7) / 8;               // 8 waves / block (256 thr)
        if ((M & 15) == 0)
            gemm_wmma_kernel<true><<<blocks, 256, 0, stream>>>(A, lda, W, ldw, bias,
                                                               C, ldc, M, N, K, act, accum);
        else
            gemm_wmma_kernel<false><<<blocks, 256, 0, stream>>>(A, lda, W, ldw, bias,
                                                                C, ldc, M, N, K, act, accum);
    };
    auto f2h = [&](const float* s, _Float16* d, int n) {
        f2h_kernel<<<(n + 255) / 256, 256, 0, stream>>>(s, d, n);
    };

    // ---- phase 0: init + f16 weight conversions ----
    init_kernel<<<16, 256, 0, stream>>>(cov, lossacc, hdec_h);
    f2h(wih_f, wihf_h, 1024 * 256);
    f2h(whh_f, whhf_h, 1024 * 256);
    f2h(wih_b, wihb_h, 1024 * 256);
    f2h(whh_b, whhb_h, 1024 * 256);
    f2h(w_sel1, wsel1_h, 256 * 512);
    f2h(w_attn, wattn_h, 256 * 1024);
    f2h(wih_d, wihd_h, 1024 * 768);
    f2h(whh_d, whhd_h, 1024 * 256);
    f2h(w_gen, wgen_h, Vc * Hc);
    embed_kernel<<<(Sc * Bc * Hc + 255) / 256, 256, 0, stream>>>(x, emb_enc, emb_h);

    // ---- phase 1: batched input projections (hoisted out of recurrence) ----
    gemm(emb_h, Hc, wihf_h, Hc, nullptr, xprojf, 1024, Sc * Bc, 1024, Hc, 0, 0);
    gemm(emb_h, Hc, wihb_h, Hc, nullptr, xprojb, 1024, Sc * Bc, 1024, Hc, 0, 0);

    // ---- phase 2: persistent bidirectional LSTM (2 blocks, WMMA per step) --
    bilstm_kernel<<<2, 256, 0, stream>>>(xprojf, xprojb, whhf_h, whhb_h,
                                         bih_f, bhh_f, bih_b, bhh_b,
                                         encout, encout_h, hTbuf, cTbuf);

    // ---- phase 3: copy probs, h0/c0, step-invariant attention terms --------
    gemm(encout_h, 2 * Hc, wsel1_h, 2 * Hc, b_sel1, sel1, Hc, Bc * Sc, Hc, 2 * Hc, 1, 0);
    copyprobs_kernel<<<(Bc * Sc + 255) / 256, 256, 0, stream>>>(sel1, w_sel2, b_sel2,
                                                                x, cp, maskcp, lossacc);
    h0c0_kernel<<<(Bc * Hc + 255) / 256, 256, 0, stream>>>(hTbuf, cTbuf, w_rh, b_rh,
                                                           w_rc, b_rc, hdec_f, hdec_h, cdec);
    // e_base = enc_out @ Wa1^T  (Wa1 = w_attn[:, :512]) — constant across steps
    gemm(encout_h, 2 * Hc, wattn_h, 1024, nullptr, ebase, Hc, Bc * Sc, Hc, 2 * Hc, 0, 0);
    uv_kernel<<<1, 256, 0, stream>>>(w_attn, w_cov, b_cov, uvec, vvec);

    // ---- phase 4: decoder loop (50 steps) ----------------------------------
    for (int t = 0; t < Tm1; ++t) {
        // hWa2 = h @ Wa2^T (Wa2 = w_attn[:, 512:768])
        gemm(hdec_h, Hc, wattn_h + 512, 1024, nullptr, hWa2, Hc, Bc, Hc, Hc, 0, 0);
        attn_kernel<<<Bc, 256, 0, stream>>>(ebase, hWa2, uvec, vvec, b_attn,
                                            w_attn_v, maskcp, encout, cov, attnw, ctx);
        decin_kernel<<<(16 * 768 + 255) / 256, 256, 0, stream>>>(labels, emb_dec, ctx,
                                                                 decin_h, t);
        gemm(decin_h, 768, wihd_h, 768, nullptr, zdec, 1024, Bc, 1024, 768, 0, 0);
        gemm(hdec_h, Hc, whhd_h, Hc, nullptr, zdec, 1024, Bc, 1024, Hc, 0, 1);
        deccell_kernel<<<(Bc * Hc + 255) / 256, 256, 0, stream>>>(zdec, bih_d, bhh_d,
                                                                  hdec_f, hdec_h, cdec);
        pgen_kernel<<<1, 32, 0, stream>>>(ctx, hdec_f, w_ptr, b_ptr, pgen);
        gemm(hdec_h, Hc, wgen_h, Hc, b_gen, vlogits, Vc, Bc, Vc, Hc, 0, 0);
        vocab_write_kernel<<<Bc, 256, 0, stream>>>(vlogits, pgen, out, t);
        scatter_kernel<<<(Bc * Sc + 255) / 256, 256, 0, stream>>>(x, attnw, pgen, out, t);
    }

    // ---- phase 5: loss -----------------------------------------------------
    celoss_kernel<<<Bc * Tm1, 256, 0, stream>>>(out, labels, lossacc);
    finalloss_kernel<<<1, 32, 0, stream>>>(lossacc, out, out_size);
}
